// NetGraph_67551245631579
// MI455X (gfx1250) — compile-verified, matrix operand order checked by reading, not run on hardware
//
#include <hip/hip_runtime.h>
#include <hip/hip_bf16.h>
#include <math.h>

typedef __attribute__((ext_vector_type(16))) _Float16 v16h;
typedef __attribute__((ext_vector_type(8)))  float    v8f;
typedef __attribute__((ext_vector_type(4)))  unsigned int tdm_u4;
typedef __attribute__((ext_vector_type(8)))  int          tdm_i8;
typedef __attribute__((ext_vector_type(4)))  int          tdm_i4;

#define CGR   256   // subgraphs
#define NN    256   // nodes per subgraph
#define EE    2048  // edges per subgraph
#define FH    128   // F_IN == NHID
#define DDIM  256   // DDI
#define GNN   273   // global nodes (C + P + 1)
#define GEE   8192  // global edges
#define LF    768   // leaf feature width (6*NHID)

// ---------------------------------------------------------------------------
// utility
// ---------------------------------------------------------------------------
__global__ void k_zero(float* p, long n) {
  long i = (long)blockIdx.x * blockDim.x + threadIdx.x;
  long st = (long)gridDim.x * blockDim.x;
  for (; i < n; i += st) p[i] = 0.f;
}

__global__ void k_copy(const float* __restrict__ s, float* __restrict__ d, long n) {
  long i = (long)blockIdx.x * blockDim.x + threadIdx.x;
  long st = (long)gridDim.x * blockDim.x;
  for (; i < n; i += st) d[i] = s[i];
}

// ---------------------------------------------------------------------------
// Pack f32 weight W (KxN row-major) into per-lane WMMA B-fragment layout, f16:
//   P[ntile][kchunk][lane][16 halves], half h of VGPR v=h/2 holds
//   K = kchunk*32 + 16*(v/4) + 2*(v%4) + 8*(lane/16) + (h&1), N-col = ntile*16 + lane%16
// ---------------------------------------------------------------------------
__global__ void k_pack_w(const float* __restrict__ W, _Float16* __restrict__ P,
                         int N, int K) {
  const int KC = K / 32;
  long t = (long)blockIdx.x * blockDim.x + threadIdx.x;
  long total = (long)N * K;
  long st = (long)gridDim.x * blockDim.x;
  for (; t < total; t += st) {
    int h     = (int)(t & 15);
    int lane  = (int)((t >> 4) & 31);
    long fi   = t >> 9;                // [ntile][kchunk]
    int kchunk = (int)(fi % KC);
    int ntile  = (int)(fi / KC);
    int v  = h >> 1;
    int kk = kchunk * 32 + 16 * (v >> 2) + 2 * (v & 3) + 8 * (lane >> 4) + (h & 1);
    int nc = ntile * 16 + (lane & 15);
    P[t] = (_Float16)W[(long)kk * N + nc];
  }
}

// ---------------------------------------------------------------------------
// WMMA GEMM:  Out[b] = A[b] (MxK, f32 row-major) * W (KxN) [+bias]
//   - W pre-packed to f16 fragments (k_pack_w)
//   - packed B block for this workgroup staged to LDS by the Tensor Data Mover
//     (tensor_load_to_lds, TENSORcnt), consumed with ds_load_b128
//   - A row index clamped (no divergent loads); stores guarded by mo < M
// block = 256 threads = 8 waves; wave w owns N-tile (blockIdx.y*8 + w).
// Requires N % 128 == 0 (true for all uses: 128, 256).
// ---------------------------------------------------------------------------
template <int K>
__global__ void k_wmma_gemm(const float* __restrict__ A, const _Float16* __restrict__ Wp,
                            const float* __restrict__ bias, float* __restrict__ Out,
                            int M, int N, long strideA, long strideO) {
  constexpr int KC = K / 32;
  extern __shared__ _Float16 sB[];          // 8 * KC * 512 halves

  const int batch = blockIdx.z;
  A   += (long)batch * strideA;
  Out += (long)batch * strideO;

  const int wave = threadIdx.x >> 5;
  const int lane = threadIdx.x & 31;
  const int grp  = lane >> 4;
  const int lm   = lane & 15;
  const int mbase = blockIdx.x * 16;
  const int ngrp  = blockIdx.y;             // group of 8 N-tiles

  // ---- stage packed B fragments for N-tiles [ngrp*8, ngrp*8+8) into LDS ----
  const _Float16* gsrc = Wp + (long)ngrp * 8 * KC * 512;
  const unsigned  nelem8 = (unsigned)(8 * KC * 512 * 2) >> 3;  // 8-byte elements

#if __has_builtin(__builtin_amdgcn_tensor_load_to_lds)
  if (wave == 0) {
    unsigned long long ga = (unsigned long long)gsrc;
    tdm_u4 g0;
    g0[0] = 1u;                                   // count=1, user mode, no gather
    g0[1] = 0u;                                   // lds_addr = 0 (dynamic-LDS base)
    g0[2] = (unsigned)(ga & 0xFFFFFFFFu);         // global_addr[31:0]
    g0[3] = (unsigned)((ga >> 32) & 0x1FFFFFFu)   // global_addr[56:32]
          | (2u << 30);                           // type = 2 (image/tensor)
    tdm_i8 g1;
    g1[0] = (int)(3u << 16);                      // wg_mask=0, data_size=3 (8B)
    g1[1] = (int)((nelem8 & 0xFFFFu) << 16);      // tensor_dim0[15:0]
    g1[2] = (int)(((nelem8 >> 16) & 0xFFFFu)      // tensor_dim0[31:16]
          | (1u << 16));                          // tensor_dim1 = 1
    g1[3] = (int)((nelem8 & 0xFFFFu) << 16);      // tile_dim0 = nelem8 (<= 24576)
    g1[4] = 0;                                    // tile_dim1/2 unused
    g1[5] = (int)nelem8;                          // tensor_dim0_stride[31:0]
    g1[6] = 0;
    g1[7] = 0;
    tdm_i4 gz4 = {0, 0, 0, 0};
    tdm_i8 gz8 = {0, 0, 0, 0, 0, 0, 0, 0};
    __builtin_amdgcn_tensor_load_to_lds(g0, g1, gz4, gz4, gz8, 0);
    __builtin_amdgcn_s_wait_tensorcnt(0);
  }
#else
  {
    const unsigned* gs = (const unsigned*)gsrc;
    unsigned* ds = (unsigned*)sB;
    for (int i = threadIdx.x; i < 8 * KC * 256; i += blockDim.x) ds[i] = gs[i];
  }
#endif
  __syncthreads();

  // ---- compute ----
  const int   m   = mbase + lm;
  const int   mc  = (m < M) ? m : (M - 1);     // clamp: row M-1 always valid
  const float* Arow = A + (long)mc * K;
  const int   nc  = (ngrp * 8 + wave) * 16 + lm;

  v8f acc = {0.f, 0.f, 0.f, 0.f, 0.f, 0.f, 0.f, 0.f};

#pragma unroll 4
  for (int kc = 0; kc < KC; ++kc) {
    // A fragment: two contiguous 8-float runs at k0+8*grp and k0+16+8*grp
    const float4* s0 = reinterpret_cast<const float4*>(Arow + kc * 32 + 8 * grp);
    const float4 q0 = s0[0], q1 = s0[1], q2 = s0[4], q3 = s0[5];
    v16h a;
    a[0]  = (_Float16)q0.x;  a[1]  = (_Float16)q0.y;
    a[2]  = (_Float16)q0.z;  a[3]  = (_Float16)q0.w;
    a[4]  = (_Float16)q1.x;  a[5]  = (_Float16)q1.y;
    a[6]  = (_Float16)q1.z;  a[7]  = (_Float16)q1.w;
    a[8]  = (_Float16)q2.x;  a[9]  = (_Float16)q2.y;
    a[10] = (_Float16)q2.z;  a[11] = (_Float16)q2.w;
    a[12] = (_Float16)q3.x;  a[13] = (_Float16)q3.y;
    a[14] = (_Float16)q3.z;  a[15] = (_Float16)q3.w;

    // B fragment: one aligned 32-byte LDS read
    const v16h b = *reinterpret_cast<const v16h*>(
        sB + ((long)(wave * KC + kc) * 32 + lane) * 16);

    acc = __builtin_amdgcn_wmma_f32_16x16x32_f16(false, a, false, b,
                                                 (short)0, acc, false, false);
  }

#pragma unroll
  for (int v = 0; v < 8; ++v) {
    // f32 D layout: lanes 0-15 -> M = v, lanes 16-31 -> M = 8+v; N = lane%16
    const int mo = mbase + v + 8 * grp;
    if (mo < M) {
      float val = acc[v];
      if (bias) val += bias[nc];
      Out[(long)mo * N + nc] = val;
    }
  }
}

// ---------------------------------------------------------------------------
// GCN pieces: deg = 1 + segsum(ew, col);  agg[col] += d^-1/2[row]*ew*d^-1/2[col]*h[row]
// out = agg + h/deg + b  (optional relu).  EW==nullptr -> weight 1.0
// ---------------------------------------------------------------------------
__global__ void k_degree(const int* __restrict__ EI, const float* __restrict__ EW,
                         float* __restrict__ Deg, int n, int E, int degStride) {
  const int c = blockIdx.x;
  const int* ei = EI + (long)c * 2 * E;
  __shared__ float sdeg[320];
  for (int i = threadIdx.x; i < n; i += blockDim.x) sdeg[i] = 1.0f;
  __syncthreads();
  for (int e = threadIdx.x; e < E; e += blockDim.x) {
    int col = ei[E + e];
    float w = EW ? EW[(long)c * E + e] : 1.0f;
    atomicAdd(&sdeg[col], w);
  }
  __syncthreads();
  for (int i = threadIdx.x; i < n; i += blockDim.x)
    Deg[(long)c * degStride + i] = sdeg[i];
}

__global__ void k_scatter(const int* __restrict__ EI, const float* __restrict__ EW,
                          const float* __restrict__ Deg, const float* __restrict__ H,
                          float* __restrict__ AGG, int n, int F, int E, int degStride) {
  const int c = blockIdx.y;
  long t = (long)blockIdx.x * blockDim.x + threadIdx.x;
  long total = (long)E * F;
  if (t >= total) return;
  int e = (int)(t / F), f = (int)(t % F);
  const int* ei = EI + (long)c * 2 * E;
  int row = ei[e], col = ei[E + e];
  float w = EW ? EW[(long)c * E + e] : 1.0f;
  if (w == 0.f) return;                 // zeroed (dropped) edge
  const float* deg = Deg + (long)c * degStride;
  float norm = rsqrtf(deg[row]) * w * rsqrtf(deg[col]);
  float v = norm * H[(long)c * n * F + (long)row * F + f];
  atomicAdd(&AGG[(long)c * n * F + (long)col * F + f], v);
}

__global__ void k_gcn_finish(const float* __restrict__ AGG, const float* __restrict__ H,
                             const float* __restrict__ Deg, const float* __restrict__ bias,
                             float* __restrict__ Out, int n, int F, int degStride, int relu) {
  const int c = blockIdx.y;
  long t = (long)blockIdx.x * blockDim.x + threadIdx.x;
  long total = (long)n * F;
  if (t >= total) return;
  int i = (int)(t / F), f = (int)(t % F);
  float d = Deg[(long)c * degStride + i];
  float v = AGG[(long)c * total + t] + H[(long)c * total + t] / d + bias[f];
  if (relu) v = fmaxf(v, 0.f);
  Out[(long)c * total + t] = v;
}

// ---------------------------------------------------------------------------
// SAGPool: score = gcn(x, Wp)[:,0]; exact top-k (jax tie-break: value desc,
// index asc); x_new = x[perm]*tanh(vals); edge remap with zeroed dropped edges.
// ---------------------------------------------------------------------------
__global__ void k_score_pool(const float* __restrict__ Xg, const int* __restrict__ EI,
                             const float* __restrict__ EW, const float* __restrict__ Deg,
                             const float* __restrict__ Wp, const float* __restrict__ bp,
                             float* __restrict__ Xout, int* __restrict__ EIout,
                             float* __restrict__ EWout,
                             int n, int k, int E, int F, int degStride, int writeEdges) {
  const int c = blockIdx.x;
  const int*   ei  = EI + (long)c * 2 * E;
  const float* ew  = EW + (long)c * E;
  const float* X   = Xg + (long)c * n * F;
  const float* deg = Deg + (long)c * degStride;

  __shared__ float s_wp[128];
  __shared__ float s_hs[256];
  __shared__ float s_agg[256];
  __shared__ float s_score[256];
  __shared__ int   s_newid[256];
  __shared__ int   s_perm[256];

  for (int f = threadIdx.x; f < F; f += blockDim.x) s_wp[f] = Wp[f];
  for (int i = threadIdx.x; i < n; i += blockDim.x) s_agg[i] = 0.f;
  __syncthreads();

  for (int i = threadIdx.x; i < n; i += blockDim.x) {
    float acc = 0.f;
    for (int f = 0; f < F; ++f) acc += X[(long)i * F + f] * s_wp[f];
    s_hs[i] = acc;
  }
  __syncthreads();

  for (int e = threadIdx.x; e < E; e += blockDim.x) {
    int row = ei[e], col = ei[E + e];
    float w = ew[e];
    if (w != 0.f) {
      float norm = rsqrtf(deg[row]) * w * rsqrtf(deg[col]);
      atomicAdd(&s_agg[col], norm * s_hs[row]);
    }
  }
  __syncthreads();

  for (int i = threadIdx.x; i < n; i += blockDim.x)
    s_score[i] = s_agg[i] + s_hs[i] / deg[i] + bp[0];
  __syncthreads();

  for (int i = threadIdx.x; i < n; i += blockDim.x) {
    float si = s_score[i];
    int r = 0;
    for (int j = 0; j < n; ++j) {
      float sj = s_score[j];
      r += (sj > si) || (sj == si && j < i);
    }
    s_newid[i] = (r < k) ? r : -1;
    if (r < k) s_perm[r] = i;
  }
  __syncthreads();

  float* Xo = Xout + (long)c * k * F;
  for (long t = threadIdx.x; t < (long)k * F; t += blockDim.x) {
    int r = (int)(t / F), f = (int)(t % F);
    int orig = s_perm[r];
    Xo[t] = X[(long)orig * F + f] * tanhf(s_score[orig]);
  }

  if (writeEdges) {
    int*   eo = EIout + (long)c * 2 * E;
    float* wo = EWout + (long)c * E;
    for (int e = threadIdx.x; e < E; e += blockDim.x) {
      int r  = s_newid[ei[e]];
      int cc = s_newid[ei[E + e]];
      bool keep = (r >= 0) && (cc >= 0);
      eo[e]     = keep ? r  : 0;
      eo[E + e] = keep ? cc : 0;
      wo[e]     = keep ? ew[e] : 0.f;
    }
  }
}

// ---------------------------------------------------------------------------
__global__ void k_readout(const float* __restrict__ X, float* __restrict__ leaf,
                          int k, int F, int off) {
  const int c = blockIdx.x;
  const float* Xc = X + (long)c * k * F;
  for (int f = threadIdx.x; f < F; f += blockDim.x) {
    float mx = -3.402823466e+38f, sm = 0.f;
    for (int r = 0; r < k; ++r) {
      float v = Xc[(long)r * F + f];
      mx = fmaxf(mx, v);
      sm += v;
    }
    leaf[(long)c * LF + off + f]     = mx;
    leaf[(long)c * LF + off + F + f] = sm / (float)k;
  }
}

// gfeat (273x768): rows 0..255 = leaf; 256..271 = parent mean of 16; 272 = global mean
__global__ void k_build_gfeat(const float* __restrict__ leaf, float* __restrict__ gfeat) {
  long t = (long)blockIdx.x * blockDim.x + threadIdx.x;
  long total = (long)GNN * LF;
  if (t >= total) return;
  int r = (int)(t / LF), f = (int)(t % LF);
  float v;
  if (r < 256) {
    v = leaf[(long)r * LF + f];
  } else if (r < 272) {
    int p = r - 256;
    float s = 0.f;
    for (int b = 0; b < 16; ++b) s += leaf[(long)(p * 16 + b) * LF + f];
    v = s / 16.f;
  } else {
    float s = 0.f;
    for (int l = 0; l < 256; ++l) s += leaf[(long)l * LF + f];
    v = s / 256.f;
  }
  gfeat[t] = v;
}

// ---------------------------------------------------------------------------
// link loss: one wave per edge; pos: softplus(-dot), neg: softplus(dot); mean.
// ---------------------------------------------------------------------------
__global__ void k_link_loss(const float* __restrict__ Y1, const float* __restrict__ Y2,
                            const int* __restrict__ ei, float* __restrict__ lossAcc,
                            int nE, int D, int isNeg) {
  int gt = blockIdx.x * blockDim.x + threadIdx.x;
  int wave = gt >> 5, lane = gt & 31;
  if (wave >= nE) return;
  int s = ei[wave], t = ei[nE + wave];
  float acc = 0.f;
  for (int d = lane; d < D; d += 32)
    acc += Y1[(long)s * D + d] * Y2[(long)t * D + d];
#pragma unroll
  for (int o = 16; o > 0; o >>= 1) acc += __shfl_xor(acc, o, 32);
  if (lane == 0) {
    float z = isNeg ? acc : -acc;
    float sp = log1pf(expf(-fabsf(z))) + fmaxf(z, 0.f);
    atomicAdd(lossAcc, sp / (float)nE);
  }
}

// ---------------------------------------------------------------------------
extern "C" void kernel_launch(void* const* d_in, const int* in_sizes, int n_in,
                              void* d_out, int out_size, void* d_ws, size_t ws_size,
                              hipStream_t stream) {
  const float* x   = (const float*)d_in[0];
  const int*   EI0 = (const int*)  d_in[1];
  const float* EW0 = (const float*)d_in[2];
  const int*   GEI = (const int*)  d_in[3];
  const int*   NEI = (const int*)  d_in[4];
  const float* W1  = (const float*)d_in[5];   const float* b1  = (const float*)d_in[6];
  const float* Wp1 = (const float*)d_in[7];   const float* bp1 = (const float*)d_in[8];
  const float* W2  = (const float*)d_in[9];   const float* b2  = (const float*)d_in[10];
  const float* Wp2 = (const float*)d_in[11];  const float* bp2 = (const float*)d_in[12];
  const float* W3  = (const float*)d_in[13];  const float* b3  = (const float*)d_in[14];
  const float* Wp3 = (const float*)d_in[15];  const float* bp3 = (const float*)d_in[16];
  const float* Wg  = (const float*)d_in[17];  const float* bg  = (const float*)d_in[18];
  const float* Wl1 = (const float*)d_in[19];  const float* bl1 = (const float*)d_in[20];
  const float* Wl2 = (const float*)d_in[21];  const float* bl2 = (const float*)d_in[22];
  float* out = (float*)d_out;

  // workspace carve-up (float units)
  float* ws = (float*)d_ws;
  long off = 0;
  auto alloc = [&](long nf) { float* p = ws + off; off += nf; return p; };
  float* Hlin = alloc((long)CGR * NN * FH);
  float* AGG  = alloc((long)CGR * NN * FH);
  float* Xg   = alloc((long)CGR * NN * FH);
  float* X1   = alloc((long)CGR * 128 * FH);
  float* X2   = alloc((long)CGR * 64 * FH);
  float* X3   = alloc((long)CGR * 32 * FH);
  float* Deg  = alloc((long)CGR * 256);
  int*   EI1  = (int*)alloc((long)CGR * 2 * EE);
  float* EW1  = alloc((long)CGR * EE);
  int*   EI2  = (int*)alloc((long)CGR * 2 * EE);
  float* EW2  = alloc((long)CGR * EE);
  float* leaf = alloc((long)CGR * LF);
  float* gfeat= alloc((long)GNN * LF);
  float* Hg   = alloc(288L * DDIM);
  float* AGGg = alloc(288L * DDIM);
  float* Degg = alloc(320);
  float* gx   = alloc(288L * DDIM);
  float* Y1   = alloc(288L * DDIM);
  float* Y2   = alloc(288L * DDIM);
  // packed f16 weights (halves; 2 halves per float slot)
  _Float16* P1  = (_Float16*)alloc(128L * 128 / 2);
  _Float16* P2  = (_Float16*)alloc(128L * 128 / 2);
  _Float16* P3  = (_Float16*)alloc(128L * 128 / 2);
  _Float16* Pg  = (_Float16*)alloc(768L * 256 / 2);
  _Float16* Pl1 = (_Float16*)alloc(256L * 256 / 2);
  _Float16* Pl2 = (_Float16*)alloc(256L * 256 / 2);
  (void)ws_size; (void)n_in; (void)in_sizes; (void)out_size;

  // pre-pack all weights into WMMA fragment layout
  k_pack_w<<<64,  256, 0, stream>>>(W1,  P1,  FH,   FH);
  k_pack_w<<<64,  256, 0, stream>>>(W2,  P2,  FH,   FH);
  k_pack_w<<<64,  256, 0, stream>>>(W3,  P3,  FH,   FH);
  k_pack_w<<<768, 256, 0, stream>>>(Wg,  Pg,  DDIM, LF);
  k_pack_w<<<256, 256, 0, stream>>>(Wl1, Pl1, DDIM, DDIM);
  k_pack_w<<<256, 256, 0, stream>>>(Wl2, Pl2, DDIM, DDIM);

  const size_t shm128 = 8 * (FH   / 32) * 512 * sizeof(_Float16);  //  32 KB
  const size_t shm256 = 8 * (DDIM / 32) * 512 * sizeof(_Float16);  //  64 KB
  const size_t shm768 = 8 * (LF   / 32) * 512 * sizeof(_Float16);  // 192 KB

  auto run_stage = [&](const float* Xin, const int* EIin, const float* EWin,
                       int n, int kk, const _Float16* Pw, const float* b,
                       const float* Wp, const float* bp,
                       float* Xout, int* EIout, float* EWout,
                       int writeEdges, int leafOff) {
    dim3 gGrid(n / 16, 1, CGR);
    k_wmma_gemm<FH><<<gGrid, 256, shm128, stream>>>(Xin, Pw, nullptr, Hlin,
                                                    n, FH, (long)n * FH, (long)n * FH);
    k_zero<<<2048, 256, 0, stream>>>(AGG, (long)CGR * n * FH);
    k_degree<<<CGR, 256, 0, stream>>>(EIin, EWin, Deg, n, EE, 256);
    dim3 sGrid((EE * FH + 255) / 256, CGR);
    k_scatter<<<sGrid, 256, 0, stream>>>(EIin, EWin, Deg, Hlin, AGG, n, FH, EE, 256);
    dim3 fGrid(((long)n * FH + 255) / 256, CGR);
    k_gcn_finish<<<fGrid, 256, 0, stream>>>(AGG, Hlin, Deg, b, Xg, n, FH, 256, 1);
    k_score_pool<<<CGR, 256, 0, stream>>>(Xg, EIin, EWin, Deg, Wp, bp,
                                          Xout, EIout, EWout, n, kk, EE, FH, 256, writeEdges);
    k_readout<<<CGR, 128, 0, stream>>>(Xout, leaf, kk, FH, leafOff);
  };

  // three per-subgraph stages
  run_stage(x,  EI0, EW0, 256, 128, P1, b1, Wp1, bp1, X1, EI1, EW1, 1, 0);
  run_stage(X1, EI1, EW1, 128,  64, P2, b2, Wp2, bp2, X2, EI2, EW2, 1, 256);
  run_stage(X2, EI2, EW2,  64,  32, P3, b3, Wp3, bp3, X3, (int*)nullptr, (float*)nullptr, 0, 512);

  // global graph
  k_build_gfeat<<<((long)GNN * LF + 255) / 256, 256, 0, stream>>>(leaf, gfeat);
  dim3 ggrid((GNN + 15) / 16, 2, 1);   // 18 M-tiles x 2 N-groups
  k_wmma_gemm<LF><<<ggrid, 256, shm768, stream>>>(gfeat, Pg, nullptr, Hg, GNN, DDIM, 0, 0);
  k_zero<<<512, 256, 0, stream>>>(AGGg, (long)GNN * DDIM);
  k_degree<<<1, 256, 0, stream>>>(GEI, (const float*)nullptr, Degg, GNN, GEE, 320);
  dim3 sg(((long)GEE * DDIM + 255) / 256, 1);
  k_scatter<<<sg, 256, 0, stream>>>(GEI, (const float*)nullptr, Degg, Hg, AGGg,
                                    GNN, DDIM, GEE, 320);
  dim3 fg(((long)GNN * DDIM + 255) / 256, 1);
  k_gcn_finish<<<fg, 256, 0, stream>>>(AGGg, Hg, Degg, bg, gx, GNN, DDIM, 320, 1);

  // class embedding = gx[:256]
  k_copy<<<256, 256, 0, stream>>>(gx, out, 65536);

  // link heads
  k_wmma_gemm<DDIM><<<ggrid, 256, shm256, stream>>>(gx, Pl1, bl1, Y1, GNN, DDIM, 0, 0);
  k_wmma_gemm<DDIM><<<ggrid, 256, shm256, stream>>>(gx, Pl2, bl2, Y2, GNN, DDIM, 0, 0);

  // loss = mean softplus(-pos) + mean softplus(neg)
  k_zero<<<1, 32, 0, stream>>>(out + 65536, 1);
  int lblocks = (GEE * 32 + 255) / 256;
  k_link_loss<<<lblocks, 256, 0, stream>>>(Y1, Y2, GEI, out + 65536, GEE, DDIM, 0);
  k_link_loss<<<lblocks, 256, 0, stream>>>(Y1, Y2, NEI, out + 65536, GEE, DDIM, 1);
}
